// Block_21182778704470
// MI455X (gfx1250) — compile-verified, matrix operand order checked by reading, not run on hardware
//
#include <hip/hip_runtime.h>
#include <hip/hip_bf16.h>
#include <math.h>

// ---------------------------------------------------------------------------
// Transformer block for MI455X (gfx1250, wave32, WMMA).
// All GEMM-like work runs on v_wmma_f32_16x16x32_bf16 (deep-K bf16 pipe);
// softmax / LN / accumulation stay in f32.  Mask handling is branchless
// (additive -1e30 bias) so the attention inner loops carry no exec-mask
// control flow.
// ---------------------------------------------------------------------------

#define B_   4
#define N_   2048
#define C_   1024
#define H_   16
#define D_   64
#define HID_ 4096
#define M_   (B_ * N_)        // 8192 rows
#define QT_  (N_ / 16)        // 128 query tiles per (b,h)

#define MASK_BIAS (-1.0e30f)

typedef __bf16 bf16_t;
typedef __attribute__((ext_vector_type(16))) __bf16 v16bf;
typedef __attribute__((ext_vector_type(8)))  __bf16 v8bf;
typedef __attribute__((ext_vector_type(8)))  float  v8f;

// ---------------------------------------------------------------------------
// WMMA wrapper: D = A(16x32 bf16) * B(32x16 bf16) + C(16x16 f32)
// ---------------------------------------------------------------------------
__device__ __forceinline__ v8f wmma_bf16(v16bf a, v16bf b, v8f c) {
  return __builtin_amdgcn_wmma_f32_16x16x32_bf16(
      /*neg_a=*/false, a, /*neg_b=*/false, b,
      /*c_mod=*/(short)0, c, /*reuse_a=*/false, /*reuse_b=*/false);
}

// Fragment loader for both A and B operands of C = A * W^T where the source is
// a row-major [rows, ld] bf16 matrix and the contraction dim is contiguous.
// ISA 16-bit A layout (wave32): lane m = l&15; lanes 0-15 hold K chunks
// [0..7],[16..23]; lanes 16-31 hold [8..15],[24..31].  B mirrors with n=l&15.
__device__ __forceinline__ v16bf load_frag(const bf16_t* __restrict__ base,
                                           int ld, int row0, int k0, int lane) {
  int r   = row0 + (lane & 15);
  int kg8 = (lane >> 4) << 3;                 // 0 or 8
  const bf16_t* p = base + (size_t)r * ld + k0 + kg8;
  v8bf lo = *(const v8bf*)(p);                // K = k0+kg8   .. +8
  v8bf hi = *(const v8bf*)(p + 16);           // K = k0+16+kg8 .. +8
  return __builtin_shufflevector(lo, hi, 0, 1, 2, 3, 4, 5, 6, 7,
                                         8, 9, 10, 11, 12, 13, 14, 15);
}

// 16-lane (half-wave) reductions: rows of the C layout live across 16 lanes.
__device__ __forceinline__ float rmax16(float v) {
  for (int d = 1; d < 16; d <<= 1) v = fmaxf(v, __shfl_xor(v, d, 32));
  return v;
}
__device__ __forceinline__ float rsum16(float v) {
  for (int d = 1; d < 16; d <<= 1) v += __shfl_xor(v, d, 32);
  return v;
}

// ---------------------------------------------------------------------------
// f32 -> bf16 convert (weights)
// ---------------------------------------------------------------------------
__global__ __launch_bounds__(256)
void cvt_bf16_kernel(const float* __restrict__ in, bf16_t* __restrict__ out,
                     size_t n) {
  for (size_t i = (size_t)blockIdx.x * blockDim.x + threadIdx.x; i < n;
       i += (size_t)gridDim.x * blockDim.x)
    out[i] = (bf16_t)in[i];
}

// ---------------------------------------------------------------------------
// LayerNorm over C_ =1024, one row per block, bf16 output
// ---------------------------------------------------------------------------
__global__ __launch_bounds__(256)
void ln_bf16_kernel(const float* __restrict__ x, const float* __restrict__ w,
                    const float* __restrict__ b, bf16_t* __restrict__ out) {
  __shared__ float red[256];
  const int row = blockIdx.x;
  const float* xr = x + (size_t)row * C_;
  float s = 0.f, s2 = 0.f;
  for (int i = threadIdx.x; i < C_; i += 256) {
    float v = xr[i];
    s += v; s2 += v * v;
  }
  red[threadIdx.x] = s; __syncthreads();
  for (int off = 128; off; off >>= 1) {
    if (threadIdx.x < off) red[threadIdx.x] += red[threadIdx.x + off];
    __syncthreads();
  }
  const float mean = red[0] * (1.0f / C_);
  __syncthreads();
  red[threadIdx.x] = s2; __syncthreads();
  for (int off = 128; off; off >>= 1) {
    if (threadIdx.x < off) red[threadIdx.x] += red[threadIdx.x + off];
    __syncthreads();
  }
  const float var = red[0] * (1.0f / C_) - mean * mean;
  const float inv = rsqrtf(var + 1e-5f);
  for (int i = threadIdx.x; i < C_; i += 256) {
    float v = (xr[i] - mean) * inv * w[i] + b[i];
    out[(size_t)row * C_ + i] = (bf16_t)v;
  }
}

// ---------------------------------------------------------------------------
// Tiled WMMA GEMM:  C[M,Nw] = A[M,K](bf16) * W[Nw,K](bf16)^T  + epilogue
//   EPI 0: store bf16
//   EPI 1: f32 out = resid + acc + bias[n]
//   EPI 2: bf16 out = gelu_exact(acc)
//   EPI 3: f32 out = resid + acc
// Block: 256 threads = 8 waves (4x2), block tile 128x64, wave tile 32x32.
// ---------------------------------------------------------------------------
template <int EPI>
__global__ __launch_bounds__(256)
void gemm_bf16_kernel(const bf16_t* __restrict__ A, const bf16_t* __restrict__ W,
                      const float* __restrict__ bias,
                      const float* __restrict__ resid,
                      void* __restrict__ outp, int Nw, int K) {
  const int lane = threadIdx.x & 31;
  const int wave = threadIdx.x >> 5;
  const int mBase = blockIdx.y * 128 + (wave >> 1) * 32;
  const int nBase = blockIdx.x * 64 + (wave & 1) * 32;

  v8f c00 = {}, c01 = {}, c10 = {}, c11 = {};

  for (int k = 0; k < K; k += 32) {
    // hint: pull next A strip toward L2 (global_prefetch_b8)
    if (k + 128 < K)
      __builtin_prefetch(A + (size_t)(mBase + (lane & 15)) * K + k + 128, 0, 1);
    v16bf a0 = load_frag(A, K, mBase,      k, lane);
    v16bf a1 = load_frag(A, K, mBase + 16, k, lane);
    v16bf b0 = load_frag(W, K, nBase,      k, lane);
    v16bf b1 = load_frag(W, K, nBase + 16, k, lane);
    c00 = wmma_bf16(a0, b0, c00);
    c01 = wmma_bf16(a0, b1, c01);
    c10 = wmma_bf16(a1, b0, c10);
    c11 = wmma_bf16(a1, b1, c11);
  }

  // C layout: VGPR j -> M = j + 8*(lane>=16); N = lane&15
  const int mo = (lane >> 4) << 3;
  const int n0 = lane & 15;
#pragma unroll
  for (int t = 0; t < 4; ++t) {
    const v8f& c = (t == 0) ? c00 : (t == 1) ? c01 : (t == 2) ? c10 : c11;
    const int mB = mBase + ((t >> 1) << 4);
    const int nB = nBase + ((t & 1) << 4);
#pragma unroll
    for (int j = 0; j < 8; ++j) {
      const int m = mB + mo + j;
      const int n = nB + n0;
      const size_t idx = (size_t)m * Nw + n;
      float v = c[j];
      if (EPI == 0) {
        ((bf16_t*)outp)[idx] = (bf16_t)v;
      } else if (EPI == 1) {
        ((float*)outp)[idx] = resid[idx] + v + bias[n];
      } else if (EPI == 2) {
        float g = 0.5f * v * (1.0f + erff(v * 0.70710678118654752f));
        ((bf16_t*)outp)[idx] = (bf16_t)g;
      } else {
        ((float*)outp)[idx] = resid[idx] + v;
      }
    }
  }
}

// ---------------------------------------------------------------------------
// V transpose:  Vt[b,h,d,n] = qkv[b,n, 2, h, d]   (bf16)
// ---------------------------------------------------------------------------
__global__ __launch_bounds__(256)
void transpose_v_kernel(const bf16_t* __restrict__ qkv, bf16_t* __restrict__ vt) {
  const size_t total = (size_t)B_ * H_ * D_ * N_;
  size_t i = (size_t)blockIdx.x * blockDim.x + threadIdx.x;
  if (i >= total) return;
  int n = (int)(i % N_);
  size_t r = i / N_;
  int d = (int)(r % D_);
  size_t r2 = r / D_;
  int h = (int)(r2 % H_);
  int b = (int)(r2 / H_);
  vt[i] = qkv[((size_t)(b * N_ + n)) * (3 * C_) + 2 * C_ + h * D_ + d];
}

// ---------------------------------------------------------------------------
// Attention: one wave per (b, h, 16-query-row tile).
//   Pass 1: S = scale*QK^T + maskbias (branchless), raw scores -> attn buffer,
//           track row max via 16-lane shuffles.
//   Pass 2: e = exp(s - max) -> attn buffer; O += P V (LDS bounce D->A layout)
//   Row-sum reciprocal -> rinv buffer; O stored normalized (bf16).
// ---------------------------------------------------------------------------
__global__ __launch_bounds__(32)
void attn_kernel(const bf16_t* __restrict__ qkv, const bf16_t* __restrict__ vt,
                 const int* __restrict__ pad_mask, const int* __restrict__ temp_p,
                 float* __restrict__ attn, bf16_t* __restrict__ o_bf16,
                 float* __restrict__ rinv_buf) {
  __shared__ __align__(16) bf16_t Pt[16 * 32];   // 16 rows x 32 keys, bf16

  const int lane = threadIdx.x;
  const int tile = blockIdx.x;                   // (b*H + h)*QT_ + qt
  const int qt = tile & (QT_ - 1);
  const int bh = tile / QT_;
  const int h = bh & (H_ - 1);
  const int b = bh >> 4;
  const int q0 = qt * 16;

  const float temp = (float)temp_p[0];
  const float scale = 0.125f / temp;             // D^-0.5 = 1/8

  // Q fragments (A layout), K-dim = D = 64 -> two 32-deep chunks
  const v16bf qa0 = load_frag(qkv, 3 * C_, b * N_ + q0, h * D_ + 0,  lane);
  const v16bf qa1 = load_frag(qkv, 3 * C_, b * N_ + q0, h * D_ + 32, lane);

  const int mo = (lane >> 4) << 3;
  const int n0 = lane & 15;

  // Branchless mask bias per query row: 0 when valid, -1e30 when padded.
  float qbias[8];
#pragma unroll
  for (int j = 0; j < 8; ++j)
    qbias[j] = pad_mask[b * N_ + q0 + mo + j] ? 0.f : MASK_BIAS;

  const size_t rowBase0 = ((size_t)bh * N_ + (q0 + mo)) * N_;   // + j*N_

  float rm[8], rs[8];
#pragma unroll
  for (int j = 0; j < 8; ++j) { rm[j] = MASK_BIAS; rs[j] = 0.f; }

  // ---- pass 1: raw masked scores + row max (no branches in loop body) ----
  for (int kt = 0; kt < QT_; ++kt) {
    const v16bf kb0 = load_frag(qkv, 3 * C_, b * N_ + kt * 16,
                                C_ + h * D_ + 0,  lane);
    const v16bf kb1 = load_frag(qkv, 3 * C_, b * N_ + kt * 16,
                                C_ + h * D_ + 32, lane);
    v8f s = {};
    s = wmma_bf16(qa0, kb0, s);
    s = wmma_bf16(qa1, kb1, s);
    const float kbias =
        pad_mask[b * N_ + kt * 16 + n0] ? 0.f : MASK_BIAS;
#pragma unroll
    for (int j = 0; j < 8; ++j) {
      float v = fmaf(s[j], scale, qbias[j] + kbias);
      attn[rowBase0 + (size_t)j * N_ + kt * 16 + n0] = v;
      rm[j] = fmaxf(rm[j], v);
    }
  }
#pragma unroll
  for (int j = 0; j < 8; ++j) rm[j] = rmax16(rm[j]);

  // ---- pass 2: exp + O = P V ----
  v8f acc[4] = {{}, {}, {}, {}};
  const bf16_t* vtb = vt + (size_t)bh * D_ * N_;

  for (int step = 0; step < N_ / 32; ++step) {
#pragma unroll
    for (int sub = 0; sub < 2; ++sub) {
      const int col0 = step * 32 + sub * 16;
#pragma unroll
      for (int j = 0; j < 8; ++j) {
        const size_t idx = rowBase0 + (size_t)j * N_ + col0 + n0;
        // masked scores carry a -1e30 bias -> exp underflows to exactly 0
        float e = __expf(attn[idx] - rm[j]);
        rs[j] += e;
        attn[idx] = e;
        Pt[(mo + j) * 32 + sub * 16 + n0] = (bf16_t)e;
      }
    }
    __syncthreads();   // single-wave WG: barrier degenerates; ds stays in order

    // reload P as A fragment from LDS
    const bf16_t* pp = Pt + (lane & 15) * 32 + ((lane >> 4) << 3);
    v8bf plo = *(const v8bf*)(pp);
    v8bf phi = *(const v8bf*)(pp + 16);
    v16bf pa = __builtin_shufflevector(plo, phi, 0, 1, 2, 3, 4, 5, 6, 7,
                                       8, 9, 10, 11, 12, 13, 14, 15);
#pragma unroll
    for (int dt = 0; dt < 4; ++dt) {
      v16bf vb = load_frag(vtb, N_, dt * 16, step * 32, lane);
      acc[dt] = wmma_bf16(pa, vb, acc[dt]);
    }
    __syncthreads();
  }

#pragma unroll
  for (int j = 0; j < 8; ++j) rs[j] = rsum16(rs[j]);

  // Fully-masked rows (rm still at MASK_BIAS) renormalize to 0 (nan_to_num).
  float ri[8];
#pragma unroll
  for (int j = 0; j < 8; ++j)
    ri[j] = (rm[j] > 0.5f * MASK_BIAS && rs[j] > 0.f) ? (1.0f / rs[j]) : 0.f;

  if (n0 == 0) {
#pragma unroll
    for (int j = 0; j < 8; ++j)
      rinv_buf[(size_t)bh * N_ + q0 + mo + j] = ri[j];
  }

  // store O (normalized) back to [b, n, h, d] layout, bf16 for proj GEMM
#pragma unroll
  for (int dt = 0; dt < 4; ++dt) {
#pragma unroll
    for (int j = 0; j < 8; ++j) {
      const int m = q0 + mo + j;
      const int d = dt * 16 + n0;
      o_bf16[((size_t)(b * N_ + m)) * C_ + h * D_ + d] =
          (bf16_t)(acc[dt][j] * ri[j]);
    }
  }
}

// ---------------------------------------------------------------------------
// attn[i] *= rinv[row]   (row = i / N_)
// ---------------------------------------------------------------------------
__global__ __launch_bounds__(256)
void attn_norm_kernel(float* __restrict__ attn, const float* __restrict__ rinv) {
  const size_t total = (size_t)B_ * H_ * N_ * N_;
  for (size_t i = (size_t)blockIdx.x * blockDim.x + threadIdx.x; i < total;
       i += (size_t)gridDim.x * blockDim.x)
    attn[i] *= rinv[i / N_];
}

// ---------------------------------------------------------------------------
// Host-side launcher
// ---------------------------------------------------------------------------
extern "C" void kernel_launch(void* const* d_in, const int* in_sizes, int n_in,
                              void* d_out, int out_size, void* d_ws,
                              size_t ws_size, hipStream_t stream) {
  (void)in_sizes; (void)n_in; (void)out_size; (void)ws_size;

  const float* x      = (const float*)d_in[0];
  const int*   temp   = (const int*)  d_in[1];
  const int*   pmask  = (const int*)  d_in[2];
  const float* ln1_w  = (const float*)d_in[3];
  const float* ln1_b  = (const float*)d_in[4];
  const float* ln2_w  = (const float*)d_in[5];
  const float* ln2_b  = (const float*)d_in[6];
  const float* qkv_w  = (const float*)d_in[7];
  const float* proj_w = (const float*)d_in[8];
  const float* proj_b = (const float*)d_in[9];
  const float* fc1_w  = (const float*)d_in[10];
  const float* fc2_w  = (const float*)d_in[11];

  float* out_x = (float*)d_out;
  float* attn  = out_x + (size_t)M_ * C_;

  // -------- carve workspace --------
  char* p = (char*)d_ws;
  auto take = [&](size_t bytes) { char* q = p; p += (bytes + 255) & ~size_t(255); return q; };
  bf16_t* h1    = (bf16_t*)take((size_t)M_ * C_ * 2);          // LN1 out
  bf16_t* wqkv  = (bf16_t*)take((size_t)3 * C_ * C_ * 2);
  bf16_t* qkvb  = (bf16_t*)take((size_t)M_ * 3 * C_ * 2);      // QKV out
  bf16_t* vtb   = (bf16_t*)take((size_t)B_ * H_ * D_ * N_ * 2);
  bf16_t* ob    = (bf16_t*)take((size_t)M_ * C_ * 2);          // attn O
  float*  rinv  = (float*) take((size_t)B_ * H_ * N_ * 4);
  bf16_t* wproj = (bf16_t*)take((size_t)C_ * C_ * 2);
  float*  x1    = (float*) take((size_t)M_ * C_ * 4);          // post-attn resid
  bf16_t* h2    = (bf16_t*)take((size_t)M_ * C_ * 2);          // LN2 out
  bf16_t* wfc1  = (bf16_t*)take((size_t)HID_ * C_ * 2);
  bf16_t* g     = (bf16_t*)take((size_t)M_ * HID_ * 2);        // GELU(fc1)
  bf16_t* wfc2  = (bf16_t*)take((size_t)C_ * HID_ * 2);

  // -------- weight conversion --------
  cvt_bf16_kernel<<<4096, 256, 0, stream>>>(qkv_w,  wqkv,  (size_t)3 * C_ * C_);
  cvt_bf16_kernel<<<4096, 256, 0, stream>>>(proj_w, wproj, (size_t)C_ * C_);
  cvt_bf16_kernel<<<4096, 256, 0, stream>>>(fc1_w,  wfc1,  (size_t)HID_ * C_);
  cvt_bf16_kernel<<<4096, 256, 0, stream>>>(fc2_w,  wfc2,  (size_t)C_ * HID_);

  // -------- attention branch --------
  ln_bf16_kernel<<<M_, 256, 0, stream>>>(x, ln1_w, ln1_b, h1);

  dim3 gQKV(3 * C_ / 64, M_ / 128);
  gemm_bf16_kernel<0><<<gQKV, 256, 0, stream>>>(h1, wqkv, nullptr, nullptr,
                                                qkvb, 3 * C_, C_);

  const size_t vtot = (size_t)B_ * H_ * D_ * N_;
  transpose_v_kernel<<<(unsigned)((vtot + 255) / 256), 256, 0, stream>>>(qkvb, vtb);

  attn_kernel<<<B_ * H_ * QT_, 32, 0, stream>>>(qkvb, vtb, pmask, temp,
                                                attn, ob, rinv);
  attn_norm_kernel<<<131072, 256, 0, stream>>>(attn, rinv);

  dim3 gProj(C_ / 64, M_ / 128);
  gemm_bf16_kernel<1><<<gProj, 256, 0, stream>>>(ob, wproj, proj_b, x,
                                                 x1, C_, C_);

  // -------- MLP branch --------
  ln_bf16_kernel<<<M_, 256, 0, stream>>>(x1, ln2_w, ln2_b, h2);

  dim3 gFc1(HID_ / 64, M_ / 128);
  gemm_bf16_kernel<2><<<gFc1, 256, 0, stream>>>(h2, wfc1, nullptr, nullptr,
                                                g, HID_, C_);

  dim3 gFc2(C_ / 64, M_ / 128);
  gemm_bf16_kernel<3><<<gFc2, 256, 0, stream>>>(g, wfc2, nullptr, x1,
                                                out_x, C_, HID_);
}
